// SimplePlayerModel_335007450021
// MI455X (gfx1250) — compile-verified, matrix operand order checked by reading, not run on hardware
//
#include <hip/hip_runtime.h>
#include <hip/hip_bf16.h>

// ---------------------------------------------------------------------------
// MI455X (gfx1250, wave32) implementation of SimplePlayerModel forward.
// All GEMM-shaped math goes through v_wmma_f32_16x16x32_f16; attention tile
// staging uses the gfx1250 async data mover (global_load_async_to_lds_b128).
// ---------------------------------------------------------------------------

typedef __attribute__((ext_vector_type(16))) _Float16 v16h;
typedef __attribute__((ext_vector_type(4)))  _Float16 v4h;
typedef __attribute__((ext_vector_type(8)))  float    v8f;
typedef __attribute__((ext_vector_type(4)))  int      v4i;

#if defined(__has_builtin)
#if __has_builtin(__builtin_amdgcn_global_load_async_to_lds_b128) && \
    __has_builtin(__builtin_amdgcn_s_wait_asynccnt)
#define USE_ASYNC_COPY 1
#endif
#endif
#ifndef USE_ASYNC_COPY
#define USE_ASYNC_COPY 0
#endif

// 16B global->LDS copy: async data-mover path on gfx1250, plain copy fallback.
// ROCm 7.2 signature (probe-verified): (int4 AS1* src, int4 AS3* dst, imm, imm).
__device__ __forceinline__ void copy16_g2l(_Float16* dst, const _Float16* src) {
#if USE_ASYNC_COPY
  __builtin_amdgcn_global_load_async_to_lds_b128(
      (__attribute__((address_space(1))) v4i*)src,
      (__attribute__((address_space(3))) v4i*)dst, 0, 0);
#else
  *reinterpret_cast<uint4*>(dst) = *reinterpret_cast<const uint4*>(src);
#endif
}
__device__ __forceinline__ void async_copy_wait() {
#if USE_ASYNC_COPY
  __builtin_amdgcn_s_wait_asynccnt(0);
#endif
}

// K-offset inside a 16-half fragment per the CDNA5 16-bit A/B layout:
// VGPR v (=h/2) holds K = (v<4 ? 2v : 16+2(v-4)) + w, plus 8*(lane/16).
// Per lane this is two contiguous 8-half (16B) runs -> 2x ds_load_b128.
__device__ __forceinline__ int koff16(int h) {
  int v = h >> 1, w = h & 1;
  return ((v < 4) ? (2 * v) : (16 + 2 * (v - 4))) + w;
}

// Fragment gather from a row-major LDS tile: element [rbase + lane%16][kbase + k]
// (used for A operands and for B operands whose "column" is the tile row).
__device__ __forceinline__ v16h frag_rowmajor(const _Float16* s, int ld, int rbase, int kbase) {
  int lane = threadIdx.x & 31;
  int r = rbase + (lane & 15);
  int g8 = (lane >> 4) << 3;
  v16h f;
#pragma unroll
  for (int h = 0; h < 16; ++h) f[h] = s[r * ld + kbase + g8 + koff16(h)];
  return f;
}

__device__ __forceinline__ v8f wmma16(v16h a, v16h b, v8f c) {
  return __builtin_amdgcn_wmma_f32_16x16x32_f16(false, a, false, b, (short)0, c, false, false);
}

__device__ __forceinline__ v4h cvt4(float4 v) {
  v4h h = {(_Float16)v.x, (_Float16)v.y, (_Float16)v.z, (_Float16)v.w};
  return h;
}

// ---------------------------------------------------------------------------
// Generic fused kernel: out = [LN]( A @ W^T + bias ) [*g+beta] [+ resid]
//   A: [M x K] f32 row-major, W: [N x K] f32 row-major (torch Linear weight).
//   One block computes a 32-row M-tile across all N columns (so LN fuses).
//   8 waves: wave w -> row-tile rt = w/4 (16 rows), col group cg = w%4 (N/4 cols).
//   If out16 != null, result is written as f16 (scaled by oscale) instead.
// ---------------------------------------------------------------------------
template <int N>
__global__ void __launch_bounds__(256) gemm_kernel(
    const float* __restrict__ A, const float* __restrict__ W,
    const float* __restrict__ bias,
    const float* __restrict__ gamma, const float* __restrict__ beta,
    const float* __restrict__ resid, float* __restrict__ out,
    _Float16* __restrict__ out16, float oscale,
    int K, int do_ln) {
  constexpr int TPW = N / 64;  // 16x16 tiles per wave
  __shared__ alignas(16) _Float16 sA[32 * 32];
  __shared__ alignas(16) _Float16 sW[N * 32];
  __shared__ float sSum[32][4];
  __shared__ float sSsq[32][4];

  const int tid = threadIdx.x;
  const int lane = tid & 31;
  const int w = tid >> 5;
  const int rt = w >> 2;       // 0..1
  const int cg = w & 3;        // 0..3
  const int colbase = cg * (N / 4);
  const int m0 = blockIdx.x * 32;

  v8f zeroV = {};
  v8f acc[TPW];
#pragma unroll
  for (int t = 0; t < TPW; ++t) acc[t] = zeroV;

  for (int kk = 0; kk < K; kk += 32) {
    // stage A tile (32x32) as f16: one float4 -> packed v4h per thread
    {
      int r = tid >> 3, c4 = (tid & 7) << 2;
      float4 v = *reinterpret_cast<const float4*>(&A[(size_t)(m0 + r) * K + kk + c4]);
      *reinterpret_cast<v4h*>(&sA[r * 32 + c4]) = cvt4(v);
    }
    // stage W tile (Nx32) as f16 (compile-time trip count: N/32 iterations)
#pragma unroll
    for (int it = 0; it < N / 32; ++it) {
      int idx = it * 256 + tid;
      int n = idx >> 3, c4 = (idx & 7) << 2;
      float4 v = *reinterpret_cast<const float4*>(&W[(size_t)n * K + kk + c4]);
      *reinterpret_cast<v4h*>(&sW[n * 32 + c4]) = cvt4(v);
    }
    if (kk + 32 < K) {  // prefetch next K-chunk (global_prefetch_b8)
      __builtin_prefetch(&A[(size_t)(m0 + (tid >> 3)) * K + kk + 32], 0, 1);
      __builtin_prefetch(&W[(size_t)tid * K + kk + 32], 0, 1);
    }
    __syncthreads();
    v16h a = frag_rowmajor(sA, 32, rt * 16, 0);
#pragma unroll
    for (int t = 0; t < TPW; ++t) {
      v16h b = frag_rowmajor(sW, 32, colbase + t * 16, 0);
      acc[t] = wmma16(a, b, acc[t]);
    }
    __syncthreads();
  }

  // bias (per-column; one column per lane per tile)
#pragma unroll
  for (int t = 0; t < TPW; ++t) {
    float bv = bias ? bias[colbase + t * 16 + (lane & 15)] : 0.0f;
#pragma unroll
    for (int j = 0; j < 8; ++j) acc[t][j] += bv;
  }

  float mean[8], rstd[8];
#pragma unroll
  for (int j = 0; j < 8; ++j) { mean[j] = 0.0f; rstd[j] = 1.0f; }

  const int rloc = rt * 16 + ((lane >> 4) << 3);  // first row (of 8) this lane holds

  if (do_ln) {
    float s[8], q[8];
#pragma unroll
    for (int j = 0; j < 8; ++j) { s[j] = 0.0f; q[j] = 0.0f; }
#pragma unroll
    for (int t = 0; t < TPW; ++t)
#pragma unroll
      for (int j = 0; j < 8; ++j) { float v = acc[t][j]; s[j] += v; q[j] += v * v; }
    // reduce across the 16 lanes of each half-wave (same row set per half)
#pragma unroll
    for (int j = 0; j < 8; ++j) {
      for (int mask = 8; mask >= 1; mask >>= 1) {
        s[j] += __shfl_xor(s[j], mask, 32);
        q[j] += __shfl_xor(q[j], mask, 32);
      }
    }
    if ((lane & 15) == 0) {
#pragma unroll
      for (int j = 0; j < 8; ++j) { sSum[rloc + j][cg] = s[j]; sSsq[rloc + j][cg] = q[j]; }
    }
    __syncthreads();
#pragma unroll
    for (int j = 0; j < 8; ++j) {
      float ss = sSum[rloc + j][0] + sSum[rloc + j][1] + sSum[rloc + j][2] + sSum[rloc + j][3];
      float qq = sSsq[rloc + j][0] + sSsq[rloc + j][1] + sSsq[rloc + j][2] + sSsq[rloc + j][3];
      float m = ss / (float)N;
      float v = qq / (float)N - m * m;
      mean[j] = m;
      rstd[j] = rsqrtf(v + 1e-5f);
    }
  }

  const int rbase = m0 + rloc;
#pragma unroll
  for (int t = 0; t < TPW; ++t) {
    int n = colbase + t * 16 + (lane & 15);
    float g = do_ln ? gamma[n] : 1.0f;
    float bb = do_ln ? beta[n] : 0.0f;
#pragma unroll
    for (int j = 0; j < 8; ++j) {
      float v = acc[t][j];
      if (do_ln) v = (v - mean[j]) * rstd[j] * g + bb;
      if (resid) v += resid[(size_t)(rbase + j) * N + n];
      if (out16) out16[(size_t)(rbase + j) * N + n] = (_Float16)(v * oscale);
      else       out[(size_t)(rbase + j) * N + n] = v;
    }
  }
}

// ---------------------------------------------------------------------------
// Flash attention over the S=512 axis. One block per (n, h, s-block of 64).
// Inputs are f16 (scale 1/sqrt(dh) already folded into QP). Element (s,n,h,dd)
// lives at buf[(s*64+n)*256 + h*64 + dd]. Q/K tiles are straight byte copies
// (async data mover when available); V is staged TRANSPOSED so the P@V
// B-operand is a contiguous row-major fragment gather (2x ds_load_b128).
// ---------------------------------------------------------------------------
__global__ void __launch_bounds__(256) flash_kernel(
    const _Float16* __restrict__ QP, const _Float16* __restrict__ KP,
    const _Float16* __restrict__ VP, float* __restrict__ O) {
  __shared__ alignas(16) _Float16 sQ[64 * 64];
  __shared__ alignas(16) _Float16 sK[64 * 64];
  __shared__ alignas(16) _Float16 sVt[64 * 64];  // [dd][t]
  __shared__ float    sS[64 * 64];
  __shared__ alignas(16) _Float16 sP[64 * 64];
  __shared__ float sM[64], sL[64], sF[64];

  const int tid = threadIdx.x;
  const int lane = tid & 31;
  const int w = tid >> 5;
  const int bx = blockIdx.x;
  const int nb = bx & 63;
  const int hh = (bx >> 6) & 3;
  const int sb = bx >> 8;

  const size_t rowstride = 64 * 256;                 // stride between s indices
  const size_t base = (size_t)nb * 256 + hh * 64;

  // Q tile: 2x 16B async copies per thread
#pragma unroll
  for (int it = 0; it < 2; ++it) {
    int idx = it * 256 + tid;
    int r = idx >> 3, c8 = (idx & 7) << 3;
    copy16_g2l(&sQ[r * 64 + c8], &QP[(size_t)(sb * 64 + r) * rowstride + base + c8]);
  }
  if (tid < 64) { sM[tid] = -1e30f; sL[tid] = 0.0f; }

  const int rt = w >> 1;         // 0..3 row tile of the 64x64 block
  const int ctb = (w & 1) * 2;   // first of 2 col tiles
  v8f zeroV = {};
  v8f oacc[2];
  oacc[0] = zeroV; oacc[1] = zeroV;
  const int rloc = rt * 16 + ((lane >> 4) << 3);
  async_copy_wait();
  __syncthreads();

  for (int tb = 0; tb < 8; ++tb) {
#pragma unroll
    for (int it = 0; it < 2; ++it) {                  // K tile: async copy
      int idx = it * 256 + tid;
      int r = idx >> 3, c8 = (idx & 7) << 3;
      copy16_g2l(&sK[r * 64 + c8],
                 &KP[(size_t)(tb * 64 + r) * rowstride + base + c8]);
    }
#pragma unroll
    for (int it = 0; it < 4; ++it) {                  // V tile: transposed scatter
      int idx = it * 256 + tid;
      int r = idx >> 4, c4 = (idx & 15) << 2;
      v4h vv = *reinterpret_cast<const v4h*>(
          &VP[(size_t)(tb * 64 + r) * rowstride + base + c4]);
      sVt[(c4 + 0) * 64 + r] = vv[0];
      sVt[(c4 + 1) * 64 + r] = vv[1];
      sVt[(c4 + 2) * 64 + r] = vv[2];
      sVt[(c4 + 3) * 64 + r] = vv[3];
    }
    async_copy_wait();
    __syncthreads();

    // S = Q @ K^T for this 64x64 block (B column = K-tile row -> rowmajor gather)
#pragma unroll
    for (int t = 0; t < 2; ++t) {
      v8f sacc = zeroV;
#pragma unroll
      for (int kk = 0; kk < 64; kk += 32) {
        v16h a = frag_rowmajor(sQ, 64, rt * 16, kk);
        v16h b = frag_rowmajor(sK, 64, (ctb + t) * 16, kk);
        sacc = wmma16(a, b, sacc);
      }
      int c0 = (ctb + t) * 16 + (lane & 15);
#pragma unroll
      for (int j = 0; j < 8; ++j) sS[(rloc + j) * 64 + c0] = sacc[j];
    }
    __syncthreads();

    // online softmax; wave w owns rows w*8 .. w*8+7
    for (int rr = 0; rr < 8; ++rr) {
      int row = w * 8 + rr;
      float s0 = sS[row * 64 + lane];
      float s1 = sS[row * 64 + 32 + lane];
      float mx = fmaxf(s0, s1);
      for (int mask = 16; mask >= 1; mask >>= 1) mx = fmaxf(mx, __shfl_xor(mx, mask, 32));
      float mold = sM[row];
      float mnew = fmaxf(mold, mx);
      float p0 = __expf(s0 - mnew);
      float p1 = __expf(s1 - mnew);
      float ls = p0 + p1;
      for (int mask = 16; mask >= 1; mask >>= 1) ls += __shfl_xor(ls, mask, 32);
      if (lane == 0) {
        float f = __expf(mold - mnew);
        sM[row] = mnew;
        sL[row] = sL[row] * f + ls;
        sF[row] = f;
      }
      sP[row * 64 + lane] = (_Float16)p0;
      sP[row * 64 + 32 + lane] = (_Float16)p1;
    }
    __syncthreads();

    // rescale running O, then O += P @ V   (B from transposed V: rowmajor gather)
#pragma unroll
    for (int t = 0; t < 2; ++t)
#pragma unroll
      for (int j = 0; j < 8; ++j) oacc[t][j] *= sF[rloc + j];
#pragma unroll
    for (int kk = 0; kk < 64; kk += 32) {
      v16h a = frag_rowmajor(sP, 64, rt * 16, kk);
#pragma unroll
      for (int t = 0; t < 2; ++t) {
        v16h b = frag_rowmajor(sVt, 64, (ctb + t) * 16, kk);
        oacc[t] = wmma16(a, b, oacc[t]);
      }
    }
    __syncthreads();
  }

#pragma unroll
  for (int t = 0; t < 2; ++t) {
    int c0 = (ctb + t) * 16 + (lane & 15);
#pragma unroll
    for (int j = 0; j < 8; ++j) {
      float v = oacc[t][j] / sL[rloc + j];
      O[(size_t)(sb * 64 + rloc + j) * rowstride + base + c0] = v;
    }
  }
}

// ---------------------------------------------------------------------------
// 256x256 f32 transpose (for composing the chained projection weights).
// ---------------------------------------------------------------------------
__global__ void transpose256_kernel(const float* __restrict__ src, float* __restrict__ dst) {
  int r = blockIdx.x, c = threadIdx.x;
  dst[(size_t)c * 256 + r] = src[(size_t)r * 256 + c];
}

// ---------------------------------------------------------------------------
// beff = W @ vin + badd  (W: [256 x 256]); one wave per output row.
// ---------------------------------------------------------------------------
__global__ void __launch_bounds__(256) matvec_bias_kernel(
    const float* __restrict__ W, const float* __restrict__ vin,
    const float* __restrict__ badd, float* __restrict__ outb) {
  int w = threadIdx.x >> 5, lane = threadIdx.x & 31;
  int row = blockIdx.x * 8 + w;
  float acc = 0.0f;
#pragma unroll
  for (int it = 0; it < 8; ++it) {
    int c = it * 32 + lane;
    acc += W[(size_t)row * 256 + c] * vin[c];
  }
  for (int mask = 16; mask >= 1; mask >>= 1) acc += __shfl_xor(acc, mask, 32);
  if (lane == 0) outb[row] = acc + badd[row];
}

// ---------------------------------------------------------------------------
// Masked segment reduce over the player axis + concat(home, home) (bug kept).
// ---------------------------------------------------------------------------
__global__ void __launch_bounds__(256) segreduce_kernel(
    const float* __restrict__ p, const float* __restrict__ x, float* __restrict__ teams) {
  int b = blockIdx.x;
  int d = threadIdx.x;  // 256 threads
  float s = 0.0f;
  for (int pp = 0; pp < 64; ++pp) {
    float m = x[(size_t)(b * 64 + pp) * 64 + 63];
    if (m == 1.0f) s += p[(size_t)(b * 64 + pp) * 256 + d];
  }
  teams[(size_t)b * 512 + d] = s;
  teams[(size_t)b * 512 + 256 + d] = s;
}

// ---------------------------------------------------------------------------
// Final: out[b] = dot(LN(o_row; lnf), pred_w) + pred_b. One wave per row.
// ---------------------------------------------------------------------------
__global__ void __launch_bounds__(256) final_kernel(
    const float* __restrict__ o, const float* __restrict__ lnf_g,
    const float* __restrict__ lnf_b, const float* __restrict__ pred_w,
    const float* __restrict__ pred_b, float* __restrict__ out) {
  int w = threadIdx.x >> 5, lane = threadIdx.x & 31;
  int row = blockIdx.x * 8 + w;
  const float* r = o + (size_t)row * 256;
  float s = 0.0f, q = 0.0f;
#pragma unroll
  for (int it = 0; it < 8; ++it) {
    float v = r[it * 32 + lane];
    s += v; q += v * v;
  }
  for (int mask = 16; mask >= 1; mask >>= 1) {
    s += __shfl_xor(s, mask, 32);
    q += __shfl_xor(q, mask, 32);
  }
  float mean = s * (1.0f / 256.0f);
  float rstd = rsqrtf(q * (1.0f / 256.0f) - mean * mean + 1e-5f);
  float acc = 0.0f;
#pragma unroll
  for (int it = 0; it < 8; ++it) {
    int c = it * 32 + lane;
    float v = (r[c] - mean) * rstd * lnf_g[c] + lnf_b[c];
    acc += v * pred_w[c];
  }
  for (int mask = 16; mask >= 1; mask >>= 1) acc += __shfl_xor(acc, mask, 32);
  if (lane == 0) out[row] = acc + pred_b[0];
}

// ---------------------------------------------------------------------------
extern "C" void kernel_launch(void* const* d_in, const int* in_sizes, int n_in,
                              void* d_out, int out_size, void* d_ws, size_t ws_size,
                              hipStream_t stream) {
  (void)in_sizes; (void)n_in; (void)out_size; (void)ws_size;
  const float* x        = (const float*)d_in[0];
  const float* emb_w    = (const float*)d_in[1];
  const float* emb_b    = (const float*)d_in[2];
  const float* emb_g    = (const float*)d_in[3];
  const float* emb_beta = (const float*)d_in[4];
  const float* post_w   = (const float*)d_in[5];
  const float* post_b   = (const float*)d_in[6];
  const float* post_g   = (const float*)d_in[7];
  const float* post_beta= (const float*)d_in[8];
  const float* attn_qw  = (const float*)d_in[9];
  const float* attn_qb  = (const float*)d_in[10];
  const float* attn_kw  = (const float*)d_in[11];
  const float* attn_kb  = (const float*)d_in[12];
  const float* attn_vw  = (const float*)d_in[13];
  const float* attn_vb  = (const float*)d_in[14];
  const float* attn_inw = (const float*)d_in[15];
  const float* attn_inb = (const float*)d_in[16];
  const float* attn_outw= (const float*)d_in[17];
  const float* attn_outb= (const float*)d_in[18];
  const float* attn_g   = (const float*)d_in[19];
  const float* attn_beta= (const float*)d_in[20];
  const float* player_w = (const float*)d_in[21];
  const float* player_b = (const float*)d_in[22];
  const float* player_g = (const float*)d_in[23];
  const float* player_beta = (const float*)d_in[24];
  const float* team_w   = (const float*)d_in[25];
  const float* team_b   = (const float*)d_in[26];
  const float* team_g   = (const float*)d_in[27];
  const float* team_beta= (const float*)d_in[28];
  const float* pre_w    = (const float*)d_in[29];
  const float* pre_b    = (const float*)d_in[30];
  const float* pre_g    = (const float*)d_in[31];
  const float* pre_beta = (const float*)d_in[32];
  const float* lnf_g    = (const float*)d_in[33];
  const float* lnf_b    = (const float*)d_in[34];
  const float* pred_w   = (const float*)d_in[35];
  const float* pred_b   = (const float*)d_in[36];

  const size_t big = (size_t)32768 * 256;  // one activation tensor (elements)
  float*     p     = (float*)d_ws;
  float*     Ob    = p + big;
  _Float16*  QPh   = (_Float16*)(Ob + big);
  _Float16*  KPh   = QPh + big;
  _Float16*  VPh   = KPh + big;
  float*     teams = (float*)(VPh + big);            // 512 x 512
  float*     preo  = teams + (size_t)512 * 512;      // 512 x 256
  float*     Tbuf  = preo + (size_t)512 * 256;       // 256 x 256 transpose scratch
  float*     Wq    = Tbuf + (size_t)256 * 256;       // composed weights
  float*     Wk    = Wq + (size_t)256 * 256;
  float*     Wv    = Wk + (size_t)256 * 256;
  float*     bqe   = Wv + (size_t)256 * 256;         // composed biases (3 x 256)
  float*     bke   = bqe + 256;
  float*     bve   = bke + 256;

  dim3 blk(256);
  const int gb = 32768 / 32;   // M-tiles for token GEMMs
  const int gt = 512 / 32;     // M-tiles for team-level GEMMs
  const int gw = 256 / 32;     // M-tiles for 256-row weight-compose GEMMs

  // embedding: p = LN(x @ emb_w^T + b)
  gemm_kernel<256><<<gb, blk, 0, stream>>>(x, emb_w, emb_b, emb_g, emb_beta,
                                           nullptr, p, nullptr, 1.0f, 64, 1);
  // post_embedding: p = LN(p @ post_w^T + b) + p
  gemm_kernel<256><<<gb, blk, 0, stream>>>(p, post_w, post_b, post_g, post_beta,
                                           p, p, nullptr, 1.0f, 256, 1);

  for (int i = 0; i < 2; ++i) {
    const float* qw = attn_qw + (size_t)i * 256 * 256;
    const float* kw = attn_kw + (size_t)i * 256 * 256;
    const float* vw = attn_vw + (size_t)i * 256 * 256;
    const float* inw = attn_inw + (size_t)i * 3 * 256 * 256;
    const float* inb = attn_inb + (size_t)i * 3 * 256;

    // Compose the chained projections: W_eff = wq @ aqw, b_eff = wq@aqb + bq.
    transpose256_kernel<<<256, blk, 0, stream>>>(qw, Tbuf);
    gemm_kernel<256><<<gw, blk, 0, stream>>>(inw, Tbuf, nullptr, nullptr, nullptr,
                                             nullptr, Wq, nullptr, 1.0f, 256, 0);
    matvec_bias_kernel<<<32, blk, 0, stream>>>(inw, attn_qb + i * 256, inb, bqe);

    transpose256_kernel<<<256, blk, 0, stream>>>(kw, Tbuf);
    gemm_kernel<256><<<gw, blk, 0, stream>>>(inw + 256 * 256, Tbuf, nullptr, nullptr, nullptr,
                                             nullptr, Wk, nullptr, 1.0f, 256, 0);
    matvec_bias_kernel<<<32, blk, 0, stream>>>(inw + 256 * 256, attn_kb + i * 256,
                                               inb + 256, bke);

    transpose256_kernel<<<256, blk, 0, stream>>>(vw, Tbuf);
    gemm_kernel<256><<<gw, blk, 0, stream>>>(inw + 2 * 256 * 256, Tbuf, nullptr, nullptr, nullptr,
                                             nullptr, Wv, nullptr, 1.0f, 256, 0);
    matvec_bias_kernel<<<32, blk, 0, stream>>>(inw + 2 * 256 * 256, attn_vb + i * 256,
                                               inb + 512, bve);

    // Single-pass projections straight to f16 (Q carries the 1/sqrt(dh) scale).
    gemm_kernel<256><<<gb, blk, 0, stream>>>(p, Wq, bqe, nullptr, nullptr,
                                             nullptr, nullptr, QPh, 0.125f, 256, 0);
    gemm_kernel<256><<<gb, blk, 0, stream>>>(p, Wk, bke, nullptr, nullptr,
                                             nullptr, nullptr, KPh, 1.0f, 256, 0);
    gemm_kernel<256><<<gb, blk, 0, stream>>>(p, Wv, bve, nullptr, nullptr,
                                             nullptr, nullptr, VPh, 1.0f, 256, 0);

    // attention over S=512 per (n,h)
    flash_kernel<<<64 * 4 * 8, blk, 0, stream>>>(QPh, KPh, VPh, Ob);

    // out_proj + LN + residual
    gemm_kernel<256><<<gb, blk, 0, stream>>>(Ob, attn_outw + (size_t)i * 256 * 256,
                                             attn_outb + i * 256, attn_g + i * 256,
                                             attn_beta + i * 256, p, p, nullptr, 1.0f, 256, 1);
  }

  for (int i = 0; i < 2; ++i) {
    gemm_kernel<256><<<gb, blk, 0, stream>>>(p, player_w + (size_t)i * 256 * 256,
                                             player_b + i * 256, player_g + i * 256,
                                             player_beta + i * 256, p, p, nullptr, 1.0f, 256, 1);
  }

  segreduce_kernel<<<512, blk, 0, stream>>>(p, x, teams);

  for (int i = 0; i < 2; ++i) {
    gemm_kernel<512><<<gt, blk, 0, stream>>>(teams, team_w + (size_t)i * 512 * 512,
                                             team_b + i * 512, team_g + i * 512,
                                             team_beta + i * 512, teams, teams,
                                             nullptr, 1.0f, 512, 1);
  }

  gemm_kernel<256><<<gt, blk, 0, stream>>>(teams, pre_w, pre_b, pre_g, pre_beta,
                                           nullptr, preo, nullptr, 1.0f, 512, 1);

  final_kernel<<<512 / 8, blk, 0, stream>>>(preo, lnf_g, lnf_b, pred_w, pred_b,
                                            (float*)d_out);
}
